// VQEmbeddingEMA_84052509982994
// MI455X (gfx1250) — compile-verified, matrix operand order checked by reading, not run on hardware
//
#include <hip/hip_runtime.h>
#include <hip/hip_bf16.h>

typedef __attribute__((ext_vector_type(16))) __bf16 v16bf;
typedef __attribute__((ext_vector_type(8)))  __bf16 v8bf;
typedef __attribute__((ext_vector_type(8)))  float  v8f;

#define D_DIM 512
#define K_CODES 512
#define N_ROWS 32768            // 16*2048
#define ZQ_ELEMS 16777216       // 32768*512
#define N_TILES 32              // 512 codes / 16
#define D_STEPS 16              // 512 / 32

// ---------------- prep: emb f32 -> bf16 copy, per-code squared norms ----------------
__global__ void vq_prep(const float* __restrict__ emb,
                        __bf16* __restrict__ embbf,
                        float* __restrict__ en2) {
    int k = blockIdx.x;          // code row
    int t = threadIdx.x;         // 256 threads
    float s = 0.f;
    for (int c = t; c < D_DIM; c += 256) {
        float v = emb[(size_t)k * D_DIM + c];
        embbf[(size_t)k * D_DIM + c] = (__bf16)v;
        s += v * v;
    }
    __shared__ float red[256];
    red[t] = s;
    __syncthreads();
    for (int off = 128; off > 0; off >>= 1) {
        if (t < off) red[t] += red[t + off];
        __syncthreads();
    }
    if (t == 0) en2[k] = red[0];
}

// A fragment: 16-bit A 16x32 layout. Lane L<16 -> row L, K = d..d+7 and d+16..d+23.
// Lanes 16..31 -> same rows, K offset +8 (handled by caller via hi * 8 in zrow).
__device__ __forceinline__ v16bf load_frag_f32(const float* __restrict__ base) {
    float4 f0 = *(const float4*)(base);
    float4 f1 = *(const float4*)(base + 4);
    float4 f2 = *(const float4*)(base + 16);
    float4 f3 = *(const float4*)(base + 20);
    v16bf a;
    a[0]  = (__bf16)f0.x; a[1]  = (__bf16)f0.y; a[2]  = (__bf16)f0.z; a[3]  = (__bf16)f0.w;
    a[4]  = (__bf16)f1.x; a[5]  = (__bf16)f1.y; a[6]  = (__bf16)f1.z; a[7]  = (__bf16)f1.w;
    a[8]  = (__bf16)f2.x; a[9]  = (__bf16)f2.y; a[10] = (__bf16)f2.z; a[11] = (__bf16)f2.w;
    a[12] = (__bf16)f3.x; a[13] = (__bf16)f3.y; a[14] = (__bf16)f3.z; a[15] = (__bf16)f3.w;
    return a;
}

__device__ __forceinline__ v16bf load_frag_bf16(const __bf16* __restrict__ base) {
    v8bf lo = *(const v8bf*)(base);        // 16B aligned
    v8bf hi = *(const v8bf*)(base + 16);
    v16bf r;
#pragma unroll
    for (int i = 0; i < 8; ++i) { r[i] = lo[i]; r[8 + i] = hi[i]; }
    return r;
}

// ---------------- main fused kernel ----------------
// grid: 256 blocks x 256 threads (8 waves). Wave w: rows [blk*128 + w*16, +16).
__global__ void __launch_bounds__(256, 1)
vq_main(const float* __restrict__ z,
        const float* __restrict__ emb_f32,
        const __bf16* __restrict__ embbf,
        const float* __restrict__ en2,
        float* __restrict__ out_zq,
        float* __restrict__ out_idx,
        float* __restrict__ loss_partials) {
    const int lane = threadIdx.x & 31;
    const int wave = threadIdx.x >> 5;
    const int ncol = lane & 15;           // column within 16x16 tile / A row offset
    const int hi   = lane >> 4;           // 0: K 0..7/16..23 , 1: K 8..15/24..31
    const size_t m0 = (size_t)blockIdx.x * 128 + (size_t)wave * 16;

    __shared__ int sidx[8][16];

    // code squared-norms for this lane's columns: code = n*16 + ncol
    float en2v[N_TILES];
#pragma unroll
    for (int n = 0; n < N_TILES; ++n) en2v[n] = en2[n * 16 + ncol];

    v8f acc[N_TILES];
#pragma unroll
    for (int n = 0; n < N_TILES; ++n) { v8f zz = {}; acc[n] = zz; }

    const float*  zrow  = z + (m0 + (size_t)ncol) * D_DIM + hi * 8;
    const __bf16* ebase = embbf + (size_t)ncol * D_DIM + hi * 8;

    // K-loop over D: one A frag, 32 B frags (all 512 codes), 32 WMMAs per step
#pragma unroll 1
    for (int dstep = 0; dstep < D_STEPS; ++dstep) {
        const int dbase = dstep * 32;
        v16bf afrag = load_frag_f32(zrow + dbase);
#pragma unroll
        for (int n = 0; n < N_TILES; ++n) {
            v16bf bfrag = load_frag_bf16(ebase + (size_t)n * 16 * D_DIM + dbase);
            acc[n] = __builtin_amdgcn_wmma_f32_16x16x32_bf16(
                false, afrag, false, bfrag, (short)0, acc[n], false, false);
        }
    }

    // argmin of (||e||^2 - 2 z.e) per row; C layout: lane<16 holds N=lane, M=j;
    // lanes>=16 hold N=lane-16, M=j+8.
#pragma unroll
    for (int j = 0; j < 8; ++j) {
        float best = 3.4e38f;
        int   bidx = 0x7fffffff;
#pragma unroll
        for (int n = 0; n < N_TILES; ++n) {
            float s = en2v[n] - 2.0f * acc[n][j];
            if (s < best) { best = s; bidx = n * 16 + ncol; }   // within-lane: first min wins
        }
#pragma unroll
        for (int off = 1; off < 16; off <<= 1) {                // reduce across 16-lane half
            float ov = __shfl_xor(best, off, 32);
            int   oi = __shfl_xor(bidx, off, 32);
            if (ov < best || (ov == best && oi < bidx)) { best = ov; bidx = oi; }
        }
        int r = hi * 8 + j;
        if (ncol == 0) {
            sidx[wave][r] = bidx;
            out_idx[m0 + r] = (float)bidx;
        }
    }
    __syncthreads();

    // gather z_q = emb[idx] (f32), stream out, accumulate squared error
    float lsum = 0.f;
#pragma unroll 1
    for (int r = 0; r < 16; ++r) {
        int code = sidx[wave][r];
        const float* e  = emb_f32 + (size_t)code * D_DIM;
        const float* zr = z      + (m0 + r) * D_DIM;
        float*       o  = out_zq + (m0 + r) * D_DIM;
#pragma unroll
        for (int c = 0; c < 4; ++c) {
            int col = (c * 32 + lane) * 4;
            float4 ev = *(const float4*)(e  + col);
            float4 zv = *(const float4*)(zr + col);
            *(float4*)(o + col) = ev;
            float dx = ev.x - zv.x, dy = ev.y - zv.y;
            float dz2 = ev.z - zv.z, dw = ev.w - zv.w;
            lsum += dx * dx + dy * dy + dz2 * dz2 + dw * dw;
        }
    }
#pragma unroll
    for (int off = 1; off < 32; off <<= 1) lsum += __shfl_xor(lsum, off, 32);
    if (lane == 0) loss_partials[blockIdx.x * 8 + wave] = lsum;
}

// ---------------- deterministic loss reduction ----------------
__global__ void vq_loss_reduce(const float* __restrict__ partials,
                               float* __restrict__ out_loss) {
    __shared__ float red[256];
    int t = threadIdx.x;
    float s = 0.f;
    for (int i = t; i < 2048; i += 256) s += partials[i];
    red[t] = s;
    __syncthreads();
    for (int off = 128; off > 0; off >>= 1) {
        if (t < off) red[t] += red[t + off];
        __syncthreads();
    }
    // loss = (1 + 0.25) * mean((z_q - z)^2)
    if (t == 0) out_loss[0] = red[0] * (1.25f / 16777216.0f);
}

extern "C" void kernel_launch(void* const* d_in, const int* in_sizes, int n_in,
                              void* d_out, int out_size, void* d_ws, size_t ws_size,
                              hipStream_t stream) {
    const float* z   = (const float*)d_in[0];   // [16,2048,512] f32
    const float* emb = (const float*)d_in[1];   // [512,512] f32

    // workspace layout
    __bf16* embbf = (__bf16*)d_ws;                                  // 512 KB
    float*  en2   = (float*)((char*)d_ws + (size_t)K_CODES * D_DIM * 2);
    float*  lpart = en2 + K_CODES;                                  // 2048 floats

    float* out      = (float*)d_out;
    float* out_zq   = out;                       // 16,777,216 f32
    float* out_loss = out + ZQ_ELEMS;            // 1 f32
    float* out_idx  = out + ZQ_ELEMS + 1;        // 32,768 f32 (indices as float)

    vq_prep<<<K_CODES, 256, 0, stream>>>(emb, embbf, en2);
    vq_main<<<N_ROWS / 128, 256, 0, stream>>>(z, emb, embbf, en2,
                                              out_zq, out_idx, lpart);
    vq_loss_reduce<<<1, 256, 0, stream>>>(lpart, out_loss);
}
